// HeteroGraph_75514114998319
// MI455X (gfx1250) — compile-verified, matrix operand order checked by reading, not run on hardware
//
#include <hip/hip_runtime.h>
#include <hip/hip_bf16.h>

// ---------------------------------------------------------------------------
// Types for CDNA5 WMMA (wave32, 16x16x32 bf16 -> f32)
// ---------------------------------------------------------------------------
typedef __attribute__((ext_vector_type(16))) __bf16 v16bf;
typedef __attribute__((ext_vector_type(8)))  __bf16 v8bf;
typedef __attribute__((ext_vector_type(8)))  float  v8f;

#define H512 512

// ---------------------------------------------------------------------------
// Generic weight prep: o = bf16(a [+ b + c + d])   (null pointers skipped)
// ---------------------------------------------------------------------------
__global__ void k_sumN_cvt(const float* __restrict__ a, const float* __restrict__ b,
                           const float* __restrict__ c, const float* __restrict__ d,
                           __bf16* __restrict__ o, int n) {
    int i = blockIdx.x * blockDim.x + threadIdx.x;
    if (i >= n) return;
    float s = a[i];
    if (b) s += b[i];
    if (c) s += c[i];
    if (d) s += d[i];
    o[i] = (__bf16)s;
}

__global__ void k_sumN_f32(const float* __restrict__ a, const float* __restrict__ b,
                           const float* __restrict__ c, const float* __restrict__ d,
                           float* __restrict__ o, int n) {
    int i = blockIdx.x * blockDim.x + threadIdx.x;
    if (i >= n) return;
    float s = a[i];
    if (b) s += b[i];
    if (c) s += c[i];
    if (d) s += d[i];
    o[i] = s;
}

// ---------------------------------------------------------------------------
// Encoder: h[n][j] = bf16( b[j] + sum_k x[n][k]*W[k][j] ), tiny K (2/4/32)
// ---------------------------------------------------------------------------
__global__ void k_encode(const float* __restrict__ x, const float* __restrict__ W,
                         const float* __restrict__ b, __bf16* __restrict__ h,
                         int Nn, int Kf) {
    size_t i = (size_t)blockIdx.x * blockDim.x + threadIdx.x;
    if (i >= (size_t)Nn * H512) return;
    int n = (int)(i >> 9), j = (int)(i & 511);
    float s = b[j];
    const float* xr = x + (size_t)n * Kf;
    for (int k = 0; k < Kf; ++k) s += xr[k] * W[k * H512 + j];
    h[i] = (__bf16)s;
}

// ---------------------------------------------------------------------------
// Edge scatter for segment mean: acc[dst] += f32(h_src[src]); cnt[dst]++
// one block (256 threads) per edge, 2 channels per thread
// ---------------------------------------------------------------------------
__global__ __launch_bounds__(256) void k_scatter(const __bf16* __restrict__ hsrc,
                                                 const int* __restrict__ src,
                                                 const int* __restrict__ dst,
                                                 float* __restrict__ acc,
                                                 int* __restrict__ cnt, int E) {
    int e = blockIdx.x;
    if (e >= E) return;
    int s = src[e], d = dst[e];
    const __bf16* xs = hsrc + (size_t)s * H512;
    float* ad = acc + (size_t)d * H512;
    int t = threadIdx.x;
    atomicAdd(&ad[t],       (float)xs[t]);
    atomicAdd(&ad[t + 256], (float)xs[t + 256]);
    if (t == 0) atomicAdd(&cnt[d], 1);
}

// agg[m][k] = bf16( acc[m][k] / max(cnt[m],1) )
__global__ void k_agg_finalize(const float* __restrict__ acc, const int* __restrict__ cnt,
                               __bf16* __restrict__ agg, int Nd) {
    size_t i = (size_t)blockIdx.x * blockDim.x + threadIdx.x;
    if (i >= (size_t)Nd * H512) return;
    int m = (int)(i >> 9);
    int c = cnt[m];
    float inv = 1.0f / (float)(c > 0 ? c : 1);
    agg[i] = (__bf16)(acc[i] * inv);
}

// ---------------------------------------------------------------------------
// bf16 WMMA GEMM: C[M][N] (+)= A[M][K] @ B[K][N], all row-major bf16.
// Block = 256 threads (8 wave32), 128x64 tile, K stepped by 32.
// Each wave: 32x32 subtile = 2x2 v_wmma_f32_16x16x32_bf16 tiles.
// M % 128 == 0, N % 64 == 0, K % 32 == 0 guaranteed by caller.
// ---------------------------------------------------------------------------
__global__ __launch_bounds__(256) void k_gemm_bf16(const __bf16* __restrict__ A,
                                                   const __bf16* __restrict__ B,
                                                   __bf16* __restrict__ C,
                                                   int M, int K, int N, int accumulate) {
    __shared__ __align__(16) __bf16 As[128][40];  // [m][k], padded row (80B)
    __shared__ __align__(16) __bf16 Bs[64][40];   // [n][k] (transposed), padded row

    const int t    = threadIdx.x;
    const int wave = t >> 5;
    const int lane = t & 31;
    const int l16  = lane & 15;
    const int half = lane >> 4;
    const int waveM = (wave & 3) * 32;   // 0,32,64,96
    const int waveN = (wave >> 2) * 32;  // 0,32

    const int bm = blockIdx.x * 128;
    const int bn = blockIdx.y * 64;

    v8f acc[2][2];
#pragma unroll
    for (int i = 0; i < 2; ++i)
#pragma unroll
        for (int j = 0; j < 2; ++j)
            acc[i][j] = (v8f){0.f, 0.f, 0.f, 0.f, 0.f, 0.f, 0.f, 0.f};

    for (int k0 = 0; k0 < K; k0 += 32) {
        // ---- stage A tile: 128x32 bf16, 8-byte chunks (4 elems) ----
#pragma unroll
        for (int c = t; c < 1024; c += 256) {
            int m  = c >> 3;
            int k4 = (c & 7) << 2;
            unsigned long long v =
                *(const unsigned long long*)(A + (size_t)(bm + m) * K + k0 + k4);
            *(unsigned long long*)(&As[m][k4]) = v;
        }
        // ---- stage B tile transposed: 32(k) x 64(n) -> Bs[n][k] ----
#pragma unroll
        for (int c = t; c < 512; c += 256) {
            int kk = c >> 4;
            int n4 = (c & 15) << 2;
            const __bf16* sp = B + (size_t)(k0 + kk) * N + bn + n4;
            __bf16 b0 = sp[0], b1 = sp[1], b2 = sp[2], b3 = sp[3];
            Bs[n4 + 0][kk] = b0;
            Bs[n4 + 1][kk] = b1;
            Bs[n4 + 2][kk] = b2;
            Bs[n4 + 3][kk] = b3;
        }
        __syncthreads();

        // ---- build fragments (contiguous 16B LDS reads) ----
        v16bf afrag[2], bfrag[2];
#pragma unroll
        for (int i = 0; i < 2; ++i) {
            int row = waveM + i * 16 + l16;
            v8bf lo = *(const v8bf*)&As[row][half * 8];
            v8bf hi = *(const v8bf*)&As[row][half * 8 + 16];
            v16bf f;
#pragma unroll
            for (int e = 0; e < 8; ++e) { f[e] = lo[e]; f[e + 8] = hi[e]; }
            afrag[i] = f;
        }
#pragma unroll
        for (int j = 0; j < 2; ++j) {
            int col = waveN + j * 16 + l16;
            v8bf lo = *(const v8bf*)&Bs[col][half * 8];
            v8bf hi = *(const v8bf*)&Bs[col][half * 8 + 16];
            v16bf f;
#pragma unroll
            for (int e = 0; e < 8; ++e) { f[e] = lo[e]; f[e + 8] = hi[e]; }
            bfrag[j] = f;
        }

#pragma unroll
        for (int i = 0; i < 2; ++i)
#pragma unroll
            for (int j = 0; j < 2; ++j)
                acc[i][j] = __builtin_amdgcn_wmma_f32_16x16x32_bf16(
                    false, afrag[i], false, bfrag[j], (short)0, acc[i][j], false, false);
        __syncthreads();
    }

    // ---- epilogue: lane holds col n=l16, rows m = half*8 + r ----
#pragma unroll
    for (int i = 0; i < 2; ++i) {
#pragma unroll
        for (int j = 0; j < 2; ++j) {
            int gm0 = bm + waveM + i * 16 + half * 8;
            int gn  = bn + waveN + j * 16 + l16;
            __bf16* cp = C + (size_t)gm0 * N + gn;
            v8f v = acc[i][j];
#pragma unroll
            for (int r = 0; r < 8; ++r) {
                float x = v[r];
                if (accumulate) x += (float)cp[(size_t)r * N];
                cp[(size_t)r * N] = (__bf16)x;
            }
        }
    }
}

// ---------------------------------------------------------------------------
// In-place: h = LayerNorm( ELU( (h + blsum) * invn ) ) * g + beta, row = 512
// ---------------------------------------------------------------------------
__global__ __launch_bounds__(256) void k_elu_ln(__bf16* __restrict__ h,
                                                const float* __restrict__ blsum, float invn,
                                                const float* __restrict__ g,
                                                const float* __restrict__ beta, int Nn) {
    __shared__ float red[256];
    int row = blockIdx.x;
    if (row >= Nn) return;
    __bf16* hr = h + (size_t)row * H512;
    int t = threadIdx.x;

    float v0 = ((float)hr[t]       + blsum[t])       * invn;
    float v1 = ((float)hr[t + 256] + blsum[t + 256]) * invn;
    float e0 = v0 > 0.f ? v0 : expm1f(v0);
    float e1 = v1 > 0.f ? v1 : expm1f(v1);

    red[t] = e0 + e1;
    __syncthreads();
    for (int o = 128; o > 0; o >>= 1) {
        if (t < o) red[t] += red[t + o];
        __syncthreads();
    }
    float mean = red[0] * (1.f / 512.f);
    __syncthreads();

    float d0 = e0 - mean, d1 = e1 - mean;
    red[t] = d0 * d0 + d1 * d1;
    __syncthreads();
    for (int o = 128; o > 0; o >>= 1) {
        if (t < o) red[t] += red[t + o];
        __syncthreads();
    }
    float r = rsqrtf(red[0] * (1.f / 512.f) + 1e-5f);

    hr[t]       = (__bf16)(d0 * r * g[t]       + beta[t]);
    hr[t + 256] = (__bf16)(d1 * r * g[t + 256] + beta[t + 256]);
}

// ---------------------------------------------------------------------------
// global mean pool over operator nodes + linear head
// ---------------------------------------------------------------------------
__global__ void k_pool_add(const __bf16* __restrict__ h, const int* __restrict__ batch,
                           float* __restrict__ pool, int Nn) {
    size_t i = (size_t)blockIdx.x * blockDim.x + threadIdx.x;
    if (i >= (size_t)Nn * H512) return;
    int n = (int)(i >> 9), j = (int)(i & 511);
    atomicAdd(&pool[(size_t)batch[n] * H512 + j], (float)h[i]);
}

__global__ void k_pool_cnt(const int* __restrict__ batch, float* __restrict__ pcnt, int Nn) {
    int n = blockIdx.x * blockDim.x + threadIdx.x;
    if (n < Nn) atomicAdd(&pcnt[batch[n]], 1.0f);
}

__global__ __launch_bounds__(256) void k_head(const float* __restrict__ pool,
                                              const float* __restrict__ pcnt,
                                              const float* __restrict__ Wout,
                                              const float* __restrict__ bout,
                                              float* __restrict__ out) {
    __shared__ float red[256];
    int gidx = blockIdx.x, t = threadIdx.x;
    float inv = 1.f / fmaxf(pcnt[gidx], 1.f);
    const float* pg = pool + (size_t)gidx * H512;
    red[t] = (pg[t] * inv) * Wout[t] + (pg[t + 256] * inv) * Wout[t + 256];
    __syncthreads();
    for (int o = 128; o > 0; o >>= 1) {
        if (t < o) red[t] += red[t + o];
        __syncthreads();
    }
    if (t == 0) out[gidx] = red[0] + bout[0];
}

// ---------------------------------------------------------------------------
// Host orchestration
// ---------------------------------------------------------------------------
extern "C" void kernel_launch(void* const* d_in, const int* in_sizes, int n_in,
                              void* d_out, int out_size, void* d_ws, size_t ws_size,
                              hipStream_t stream) {
    enum { OP = 0, TAB = 1, COL = 2, PRED = 3 };
    static const int NN[4] = {131072, 32768, 131072, 65536};
    static const int relSrc[7] = {TAB, PRED, COL, COL, OP, TAB, COL};
    static const int relDst[7] = {OP, OP, OP, PRED, OP, TAB, COL};
    // dst groups
    static const int grpOp[4] = {0, 1, 2, 4};

    const float* x_feat[4] = {(const float*)d_in[0], (const float*)d_in[1],
                              (const float*)d_in[2], (const float*)d_in[3]};
    const int   xK[4] = {4, 2, 32, 4};
    const float* Wenc[4] = {(const float*)d_in[4], (const float*)d_in[6],
                            (const float*)d_in[8], (const float*)d_in[10]};
    const float* benc[4] = {(const float*)d_in[5], (const float*)d_in[7],
                            (const float*)d_in[9], (const float*)d_in[11]};

    const float* convWl[2] = {(const float*)d_in[12], (const float*)d_in[15]};
    const float* convbl[2] = {(const float*)d_in[13], (const float*)d_in[16]};
    const float* convWr[2] = {(const float*)d_in[14], (const float*)d_in[17]};
    const float* gln[2]    = {(const float*)d_in[18], (const float*)d_in[20]};
    const float* bln[2]    = {(const float*)d_in[19], (const float*)d_in[21]};
    const float* W_out = (const float*)d_in[22];
    const float* b_out = (const float*)d_in[23];

    const int* srcs[7];
    const int* dsts[7];
    int E[7];
    for (int i = 0; i < 7; ++i) {
        srcs[i] = (const int*)d_in[24 + 2 * i];
        dsts[i] = (const int*)d_in[25 + 2 * i];
        E[i] = in_sizes[24 + 2 * i];
    }
    const int* batch = (const int*)d_in[38];

    // ---- workspace carve ----
    char* wp = (char*)d_ws;
    auto carve = [&](size_t bytes) -> char* {
        char* p = wp;
        wp += (bytes + 255) & ~(size_t)255;
        return p;
    };
    __bf16* hA[4];
    __bf16* hB[4];
    for (int ty = 0; ty < 4; ++ty) hA[ty] = (__bf16*)carve((size_t)NN[ty] * H512 * 2);
    for (int ty = 0; ty < 4; ++ty) hB[ty] = (__bf16*)carve((size_t)NN[ty] * H512 * 2);
    float* acc  = (float*)carve((size_t)131072 * H512 * 4);
    int*   cnt  = (int*)carve((size_t)131072 * 4);
    __bf16* agg = (__bf16*)carve((size_t)131072 * H512 * 2);
    const size_t WB = (size_t)H512 * H512;  // elements in one 512x512
    __bf16* WlB[2][7];
    for (int l = 0; l < 2; ++l)
        for (int i = 0; i < 7; ++i) WlB[l][i] = (__bf16*)carve(WB * 2);
    __bf16* WrB[2][4];  // per dst type (OP = summed over rels 0,1,2,4)
    for (int l = 0; l < 2; ++l)
        for (int ty = 0; ty < 4; ++ty) WrB[l][ty] = (__bf16*)carve(WB * 2);
    float* blsum[2][4];
    for (int l = 0; l < 2; ++l)
        for (int ty = 0; ty < 4; ++ty) blsum[l][ty] = (float*)carve(H512 * 4);
    float* pool = (float*)carve((size_t)64 * H512 * 4);
    float* pcnt = (float*)carve(64 * 4);
    (void)ws_size; (void)n_in; (void)out_size;

    const int cvtGrid = (int)(WB / 256);

    // ---- weight prep (bf16 conversion + algebraic sums) ----
    for (int l = 0; l < 2; ++l) {
        for (int i = 0; i < 7; ++i)
            k_sumN_cvt<<<cvtGrid, 256, 0, stream>>>(convWl[l] + (size_t)i * WB,
                                                    nullptr, nullptr, nullptr,
                                                    WlB[l][i], (int)WB);
        // operator root weight: Wr0+Wr1+Wr2+Wr4
        k_sumN_cvt<<<cvtGrid, 256, 0, stream>>>(convWr[l] + 0 * WB, convWr[l] + 1 * WB,
                                                convWr[l] + 2 * WB, convWr[l] + 4 * WB,
                                                WrB[l][OP], (int)WB);
        k_sumN_cvt<<<cvtGrid, 256, 0, stream>>>(convWr[l] + 3 * WB, nullptr, nullptr, nullptr,
                                                WrB[l][PRED], (int)WB);
        k_sumN_cvt<<<cvtGrid, 256, 0, stream>>>(convWr[l] + 5 * WB, nullptr, nullptr, nullptr,
                                                WrB[l][TAB], (int)WB);
        k_sumN_cvt<<<cvtGrid, 256, 0, stream>>>(convWr[l] + 6 * WB, nullptr, nullptr, nullptr,
                                                WrB[l][COL], (int)WB);
        // bias sums per dst type
        k_sumN_f32<<<2, 256, 0, stream>>>(convbl[l] + 0 * H512, convbl[l] + 1 * H512,
                                          convbl[l] + 2 * H512, convbl[l] + 4 * H512,
                                          blsum[l][OP], H512);
        k_sumN_f32<<<2, 256, 0, stream>>>(convbl[l] + 3 * H512, nullptr, nullptr, nullptr,
                                          blsum[l][PRED], H512);
        k_sumN_f32<<<2, 256, 0, stream>>>(convbl[l] + 5 * H512, nullptr, nullptr, nullptr,
                                          blsum[l][TAB], H512);
        k_sumN_f32<<<2, 256, 0, stream>>>(convbl[l] + 6 * H512, nullptr, nullptr, nullptr,
                                          blsum[l][COL], H512);
    }

    // ---- encoders -> hA (bf16) ----
    for (int ty = 0; ty < 4; ++ty) {
        int grid = (int)(((size_t)NN[ty] * H512) / 256);
        k_encode<<<grid, 256, 0, stream>>>(x_feat[ty], Wenc[ty], benc[ty], hA[ty], NN[ty], xK[ty]);
    }

    // ---- two hetero SAGE layers ----
    for (int l = 0; l < 2; ++l) {
        __bf16** hin  = (l == 0) ? hA : hB;
        __bf16** hout = (l == 0) ? hB : hA;

        for (int ty = 0; ty < 4; ++ty) {
            const int* rels;
            int nrel;
            int single;
            if (ty == OP)        { rels = grpOp; nrel = 4; single = -1; }
            else if (ty == PRED) { single = 3; rels = &single; nrel = 1; }
            else if (ty == TAB)  { single = 5; rels = &single; nrel = 1; }
            else                 { single = 6; rels = &single; nrel = 1; }

            int Nd = NN[ty];
            dim3 ggrid(Nd / 128, H512 / 64);
            int finGrid = (int)(((size_t)Nd * H512) / 256);
            int first = 1;
            for (int r = 0; r < nrel; ++r) {
                int ri = rels[r];
                hipMemsetAsync(acc, 0, (size_t)Nd * H512 * 4, stream);
                hipMemsetAsync(cnt, 0, (size_t)Nd * 4, stream);
                k_scatter<<<E[ri], 256, 0, stream>>>(hin[relSrc[ri]], srcs[ri], dsts[ri],
                                                     acc, cnt, E[ri]);
                k_agg_finalize<<<finGrid, 256, 0, stream>>>(acc, cnt, agg, Nd);
                k_gemm_bf16<<<ggrid, 256, 0, stream>>>(agg, WlB[l][ri], hout[ty],
                                                       Nd, H512, H512, first ? 0 : 1);
                first = 0;
            }
            // root term: hin[ty] @ (sum of Wr over rels of this dst)
            k_gemm_bf16<<<ggrid, 256, 0, stream>>>(hin[ty], WrB[l][ty], hout[ty],
                                                   Nd, H512, H512, 1);
            // (x + blsum)/nrel -> ELU -> LayerNorm, in place
            k_elu_ln<<<Nd, 256, 0, stream>>>(hout[ty], blsum[l][ty], 1.0f / (float)nrel,
                                             gln[l], bln[l], Nd);
        }
    }

    // ---- global mean pool over operator nodes (layer-2 output lives in hA) + head ----
    hipMemsetAsync(pool, 0, (size_t)64 * H512 * 4, stream);
    hipMemsetAsync(pcnt, 0, 64 * 4, stream);
    {
        int grid = (int)(((size_t)NN[OP] * H512) / 256);
        k_pool_add<<<grid, 256, 0, stream>>>(hA[OP], batch, pool, NN[OP]);
        k_pool_cnt<<<NN[OP] / 256, 256, 0, stream>>>(batch, pcnt, NN[OP]);
        k_head<<<64, 256, 0, stream>>>(pool, pcnt, W_out, b_out, (float*)d_out);
    }
}
